// GraphSAGEEncoder_62517543961152
// MI455X (gfx1250) — matrix operand order repaired and checked by reading, not compiled, not run on hardware
//
#include <hip/hip_runtime.h>
#include <hip/hip_bf16.h>

typedef __attribute__((ext_vector_type(16))) __bf16 v16bf;
typedef __attribute__((ext_vector_type(8)))  __bf16 v8bf;
typedef __attribute__((ext_vector_type(8)))  float  v8f;

union FragU { v16bf v; v8bf h[2]; };

#define D    128   // hidden dim
#define DIN  256   // input dim
#define LDA  40    // A-tile row stride (elements): 80B, keeps 16B-aligned b128 frags
#define LDBI 264   // Win^T row stride (256+8): 528B, multiple of 16
#define LDW  136   // Wl^T/Wr^T row stride (128+8): 272B, multiple of 16
#define KT   32    // K per WMMA step

// ---------------------------------------------------------------- utilities
__global__ void zero_kernel(float* __restrict__ p, long n4) {
    long i = (long)blockIdx.x * blockDim.x + threadIdx.x;
    if (i < n4) ((float4*)p)[i] = float4{0.f, 0.f, 0.f, 0.f};
}

__global__ void fold_kernel(const float* __restrict__ b_in,
                            const float* __restrict__ g_in,
                            const float* __restrict__ be_in,
                            const float* __restrict__ m_in,
                            const float* __restrict__ v_in,
                            const float* __restrict__ bl,
                            const float* __restrict__ g_l,
                            const float* __restrict__ be_l,
                            const float* __restrict__ m_l,
                            const float* __restrict__ v_l,
                            float* __restrict__ fold, int L) {
    const int c = threadIdx.x;  // 128 threads
    float s = g_in[c] * rsqrtf(v_in[c] + 1e-5f);
    fold[c]       = s;
    fold[128 + c] = (b_in[c] - m_in[c]) * s + be_in[c];
    for (int l = 0; l < L; ++l) {
        float sl = g_l[l * D + c] * rsqrtf(v_l[l * D + c] + 1e-5f);
        fold[256 + l * 256 + c]       = sl;
        fold[256 + l * 256 + 128 + c] = (bl[l * D + c] - m_l[l * D + c]) * sl + be_l[l * D + c];
    }
}

__global__ void deg_kernel(const int* __restrict__ ei, float* __restrict__ deg, int E) {
    int e = blockIdx.x * blockDim.x + threadIdx.x;
    if (e < E) atomicAdd(&deg[ei[E + e]], 1.0f);
}

__global__ void finalize_deg_kernel(float* __restrict__ deg, int N) {
    int i = blockIdx.x * blockDim.x + threadIdx.x;
    if (i < N) deg[i] = 1.0f / fmaxf(deg[i], 1.0f);
}

// one wave per edge: gather h[src] row (float4/lane), scatter-add into agg[dst]
__global__ __launch_bounds__(256) void scatter_kernel(const int* __restrict__ ei,
                                                      const float* __restrict__ h,
                                                      float* __restrict__ agg, int E) {
    const int w = (int)((blockIdx.x * 256u + threadIdx.x) >> 5);
    if (w >= E) return;
    const int lane = threadIdx.x & 31;
    const int s = ei[w];
    const int d = ei[E + w];
    const float4 v = *(const float4*)(h + (size_t)s * D + lane * 4);
    float* a = agg + (size_t)d * D + lane * 4;
    atomicAdd(a + 0, v.x);
    atomicAdd(a + 1, v.y);
    atomicAdd(a + 2, v.z);
    atomicAdd(a + 3, v.w);
}

// ------------------------------------------------ input projection (WMMA)
// h = relu( x @ W_in * scale + shift )   (BN+bias folded per column)
__global__ __launch_bounds__(256) void gemm_in_kernel(const float* __restrict__ x,
                                                      const float* __restrict__ W,
                                                      const float* __restrict__ fold,
                                                      float* __restrict__ hout, int Nrows) {
    extern __shared__ char smem[];
    __bf16* sB = (__bf16*)smem;                                 // [128][LDBI] = W^T
    __bf16* sA = (__bf16*)(smem + (size_t)D * LDBI * sizeof(__bf16));  // [128][LDA]
    const int tid = threadIdx.x, lane = tid & 31, wid = tid >> 5;
    const int rowbase = blockIdx.x * 128;

    // transpose full W (256x128 f32) into LDS bf16: sB[n][k] = W[k][n]
    {
        const int k = tid;  // 0..255
        const float* wr = W + (size_t)k * D;
#pragma unroll 4
        for (int n = 0; n < D; ++n) sB[n * LDBI + k] = (__bf16)wr[n];
    }

    v8f acc[8];
#pragma unroll
    for (int t = 0; t < 8; ++t)
#pragma unroll
        for (int i = 0; i < 8; ++i) acc[t][i] = 0.0f;

    const int arow = wid * 16 + (lane & 15);
    const int kb   = (lane >> 4) * 16;   // A frag: lanes16-31 hold K+16..K+31
    const int kb2  = (lane >> 4) * 16;   // B frag: lanes16-31 hold K+16..K+31

    for (int ks = 0; ks < DIN; ks += KT) {
        __syncthreads();
        // stage 128x32 f32 tile of x as bf16 (each thread: 4 float4 loads)
#pragma unroll
        for (int i = 0; i < 4; ++i) {
            const int base = (tid + i * 256) * 4;
            const int r = base >> 5, k = base & 31;
            int grow = rowbase + r;
            if (grow >= Nrows) grow = Nrows - 1;  // clamp (stores guarded later)
            const float4 vv = *(const float4*)(x + (size_t)grow * DIN + ks + k);
            __bf16* dp = sA + r * LDA + k;
            dp[0] = (__bf16)vv.x; dp[1] = (__bf16)vv.y;
            dp[2] = (__bf16)vv.z; dp[3] = (__bf16)vv.w;
        }
        __syncthreads();
        if (ks + KT < DIN) {  // lowers to global_prefetch_b8
            const int base = tid * 4;
            const int r = base >> 5, k = base & 31;
            int grow = rowbase + r;
            if (grow >= Nrows) grow = Nrows - 1;
            __builtin_prefetch(x + (size_t)grow * DIN + ks + KT + k, 0, 0);
        }
        FragU A;
        A.h[0] = *(const v8bf*)(sA + arow * LDA + kb);        // K = kb..kb+7
        A.h[1] = *(const v8bf*)(sA + arow * LDA + kb + 8);    // K = kb+8..kb+15
#pragma unroll
        for (int t = 0; t < 8; ++t) {
            const int col = t * 16 + (lane & 15);
            FragU B;
            B.h[0] = *(const v8bf*)(sB + col * LDBI + ks + kb2);
            B.h[1] = *(const v8bf*)(sB + col * LDBI + ks + kb2 + 8);
            acc[t] = __builtin_amdgcn_wmma_f32_16x16x32_bf16(
                false, A.v, false, B.v, (short)0, acc[t], false, false);
        }
    }

    const int rsel = (lane >> 4) * 8;  // C layout: lanes16-31 are rows M+8
#pragma unroll
    for (int t = 0; t < 8; ++t) {
        const int col = t * 16 + (lane & 15);
        const float s = fold[col], b = fold[128 + col];
#pragma unroll
        for (int v = 0; v < 8; ++v) {
            const int row = rowbase + wid * 16 + rsel + v;
            if (row < Nrows) {
                float z = acc[t][v] * s + b;
                hout[(size_t)row * D + col] = z > 0.0f ? z : 0.0f;
            }
        }
    }
}

// -------------------------------------------------- SAGE layer (WMMA x2)
// hnext = hcur + relu( ((agg*rdeg) @ Wl + hcur @ Wr) * scale + shift )
__global__ __launch_bounds__(256) void gemm_layer_kernel(const float* __restrict__ agg,
                                                         const float* __restrict__ hcur,
                                                         const float* __restrict__ rdeg,
                                                         const float* __restrict__ Wl,
                                                         const float* __restrict__ Wr,
                                                         const float* __restrict__ fold,
                                                         float* __restrict__ hnext, int Nrows) {
    extern __shared__ char smem[];
    __bf16* sBl = (__bf16*)smem;             // [128][LDW] = Wl^T
    __bf16* sBr = sBl + D * LDW;             // [128][LDW] = Wr^T
    __bf16* sAa = sBr + D * LDW;             // [128][LDA]
    __bf16* sAh = sAa + 128 * LDA;           // [128][LDA]
    const int tid = threadIdx.x, lane = tid & 31, wid = tid >> 5;
    const int rowbase = blockIdx.x * 128;

    {   // threads 0..127 transpose Wl, 128..255 transpose Wr
        const int k = tid & 127;
        const float* wsrc = (tid < 128 ? Wl : Wr) + (size_t)k * D;
        __bf16* bd = (tid < 128) ? sBl : sBr;
#pragma unroll 4
        for (int n = 0; n < D; ++n) bd[n * LDW + k] = (__bf16)wsrc[n];
    }

    v8f acc[8];
#pragma unroll
    for (int t = 0; t < 8; ++t)
#pragma unroll
        for (int i = 0; i < 8; ++i) acc[t][i] = 0.0f;

    const int arow = wid * 16 + (lane & 15);
    const int kb   = (lane >> 4) * 16;
    const int kb2  = (lane >> 4) * 16;

    for (int ks = 0; ks < D; ks += KT) {
        __syncthreads();
#pragma unroll
        for (int i = 0; i < 4; ++i) {
            const int base = (tid + i * 256) * 4;
            const int r = base >> 5, k = base & 31;
            int grow = rowbase + r;
            if (grow >= Nrows) grow = Nrows - 1;
            const float rd = rdeg[grow];  // fold mean-divide into bf16 convert
            const float4 va = *(const float4*)(agg + (size_t)grow * D + ks + k);
            const float4 vh = *(const float4*)(hcur + (size_t)grow * D + ks + k);
            __bf16* da = sAa + r * LDA + k;
            da[0] = (__bf16)(va.x * rd); da[1] = (__bf16)(va.y * rd);
            da[2] = (__bf16)(va.z * rd); da[3] = (__bf16)(va.w * rd);
            __bf16* dh = sAh + r * LDA + k;
            dh[0] = (__bf16)vh.x; dh[1] = (__bf16)vh.y;
            dh[2] = (__bf16)vh.z; dh[3] = (__bf16)vh.w;
        }
        __syncthreads();
        if (ks + KT < D) {
            const int base = tid * 4;
            const int r = base >> 5, k = base & 31;
            int grow = rowbase + r;
            if (grow >= Nrows) grow = Nrows - 1;
            __builtin_prefetch(agg + (size_t)grow * D + ks + KT + k, 0, 0);
        }
        FragU Aa, Ah;
        Aa.h[0] = *(const v8bf*)(sAa + arow * LDA + kb);
        Aa.h[1] = *(const v8bf*)(sAa + arow * LDA + kb + 8);
        Ah.h[0] = *(const v8bf*)(sAh + arow * LDA + kb);
        Ah.h[1] = *(const v8bf*)(sAh + arow * LDA + kb + 8);
#pragma unroll
        for (int t = 0; t < 8; ++t) {
            const int col = t * 16 + (lane & 15);
            FragU Bm;
            Bm.h[0] = *(const v8bf*)(sBl + col * LDW + ks + kb2);
            Bm.h[1] = *(const v8bf*)(sBl + col * LDW + ks + kb2 + 8);
            acc[t] = __builtin_amdgcn_wmma_f32_16x16x32_bf16(
                false, Aa.v, false, Bm.v, (short)0, acc[t], false, false);
            Bm.h[0] = *(const v8bf*)(sBr + col * LDW + ks + kb2);
            Bm.h[1] = *(const v8bf*)(sBr + col * LDW + ks + kb2 + 8);
            acc[t] = __builtin_amdgcn_wmma_f32_16x16x32_bf16(
                false, Ah.v, false, Bm.v, (short)0, acc[t], false, false);
        }
    }

    const int rsel = (lane >> 4) * 8;
#pragma unroll
    for (int t = 0; t < 8; ++t) {
        const int col = t * 16 + (lane & 15);
        const float s = fold[col], b = fold[128 + col];
#pragma unroll
        for (int v = 0; v < 8; ++v) {
            const int row = rowbase + wid * 16 + rsel + v;
            if (row < Nrows) {
                float z = acc[t][v] * s + b;
                z = z > 0.0f ? z : 0.0f;
                hnext[(size_t)row * D + col] = z + hcur[(size_t)row * D + col];
            }
        }
    }
}

// ---------------------------------------------------------------- launcher
extern "C" void kernel_launch(void* const* d_in, const int* in_sizes, int n_in,
                              void* d_out, int out_size, void* d_ws, size_t ws_size,
                              hipStream_t stream) {
    const float* x     = (const float*)d_in[0];
    const int*   ei    = (const int*)d_in[1];
    const float* Win   = (const float*)d_in[2];
    const float* b_in  = (const float*)d_in[3];
    const float* g_in  = (const float*)d_in[4];
    const float* be_in = (const float*)d_in[5];
    const float* m_in  = (const float*)d_in[6];
    const float* v_in  = (const float*)d_in[7];
    const float* Wl    = (const float*)d_in[8];
    const float* bl    = (const float*)d_in[9];
    const float* Wr    = (const float*)d_in[10];
    const float* g_l   = (const float*)d_in[11];
    const float* be_l  = (const float*)d_in[12];
    const float* m_l   = (const float*)d_in[13];
    const float* v_l   = (const float*)d_in[14];

    const int N = in_sizes[0] / DIN;
    const int E = in_sizes[1] / 2;
    const int L = in_sizes[8] / (D * D);

    // workspace: h ping buffer | agg | rdeg | folded BN params
    char* ws = (char*)d_ws;
    float* hws  = (float*)ws;  ws += (((size_t)N * D * 4) + 255) & ~(size_t)255;
    float* agg  = (float*)ws;  ws += (((size_t)N * D * 4) + 255) & ~(size_t)255;
    float* rdeg = (float*)ws;  ws += (((size_t)N * 4) + 255) & ~(size_t)255;
    float* fold = (float*)ws;

    const int mblocks = (N + 127) / 128;
    const size_t ldsIn = (size_t)D * LDBI * 2 + (size_t)128 * LDA * 2;      // ~78 KB
    const size_t ldsLy = (size_t)2 * D * LDW * 2 + (size_t)2 * 128 * LDA * 2;  // ~88 KB

    // degrees (rdeg buffer doubles as deg accumulator) + folded BN params
    {
        long n4 = (long)N / 4;
        zero_kernel<<<(unsigned)((n4 + 255) / 256), 256, 0, stream>>>(rdeg, n4);
    }
    fold_kernel<<<1, D, 0, stream>>>(b_in, g_in, be_in, m_in, v_in,
                                     bl, g_l, be_l, m_l, v_l, fold, L);
    deg_kernel<<<(E + 255) / 256, 256, 0, stream>>>(ei, rdeg, E);
    finalize_deg_kernel<<<(N + 255) / 256, 256, 0, stream>>>(rdeg, N);

    // input projection; ping-pong so that the last layer writes d_out
    float* h0 = (L & 1) ? hws : (float*)d_out;
    gemm_in_kernel<<<mblocks, 256, ldsIn, stream>>>(x, Win, fold, h0, N);

    const float* hcur = h0;
    for (int l = 0; l < L; ++l) {
        long n4 = (long)((size_t)N * D / 4);
        zero_kernel<<<(unsigned)((n4 + 255) / 256), 256, 0, stream>>>(agg, n4);
        scatter_kernel<<<(E + 7) / 8, 256, 0, stream>>>(ei, hcur, agg, E);
        float* hn = (hcur == hws) ? (float*)d_out : hws;
        gemm_layer_kernel<<<mblocks, 256, ldsLy, stream>>>(
            agg, hcur, rdeg, Wl + (size_t)l * D * D, Wr + (size_t)l * D * D,
            fold + 256 + l * 256, hn, N);
        hcur = hn;
    }
}
